// FillSimNetAnd_472446402865
// MI455X (gfx1250) — matrix operand order repaired and checked by reading, not compiled, not run on hardware
//
#include <hip/hip_runtime.h>

typedef __bf16 bf16_t;
typedef __attribute__((ext_vector_type(16))) __bf16 v16bf;
typedef __attribute__((ext_vector_type(8)))  __bf16 v8bf;
typedef __attribute__((ext_vector_type(8)))  float  v8f;

// ---------------------------------------------------------------------------
// Degree / normalization
// ---------------------------------------------------------------------------
__global__ void k_fill_deg(float* deg, int n) {
    int i = blockIdx.x * blockDim.x + threadIdx.x;
    if (i < n) deg[i] = 1.0f;                       // self-loop weight = 1
}

__global__ void k_acc_deg(const int* __restrict__ col, const float* __restrict__ ew,
                          float* deg, int E) {
    int e = blockIdx.x * blockDim.x + threadIdx.x;
    if (e < E) atomicAdd(&deg[col[e]], ew[e]);
}

__global__ void k_deg_to_dis(float* deg, int n) {
    int i = blockIdx.x * blockDim.x + threadIdx.x;
    if (i < n) {
        float d = deg[i];
        deg[i] = (d > 0.0f) ? rsqrtf(d) : 0.0f;     // in place: deg -> dis
    }
}

__global__ void k_edge_norm(const int* __restrict__ row, const int* __restrict__ col,
                            const float* __restrict__ ew, const float* __restrict__ dis,
                            float* __restrict__ nrm, int E) {
    int e = blockIdx.x * blockDim.x + threadIdx.x;
    if (e < E) nrm[e] = dis[row[e]] * ew[e] * dis[col[e]];
}

__global__ void k_zero64(float* z) {
    if (threadIdx.x < 64) z[threadIdx.x] = 0.0f;
}

// ---------------------------------------------------------------------------
// Pack a 64x64 row-major f32 weight matrix into bf16 WMMA B-fragment order.
// Fragment f = nb*2 + kb (nb = column tile 0..3, kb = K block 0..1).
// Lane l (n = l&15, hi = l>>4), element i (0..15): B[K = kb*32 + (i>>3)*16 + hi*8 + (i&7)][nb*16 + n]
// -> each lane reads 32 contiguous bytes.
// ---------------------------------------------------------------------------
__global__ void k_pack_w(const float* __restrict__ src, bf16_t* __restrict__ dst) {
    int t = blockIdx.x * 256 + threadIdx.x;         // 4096 elements
    int f = t >> 9, lane = (t >> 4) & 31, i = t & 15;
    int kb = f & 1, nb = f >> 1;
    int K = kb * 32 + (i >> 3) * 16 + (lane >> 4) * 8 + (i & 7);
    int n = nb * 16 + (lane & 15);
    dst[t] = (bf16_t)src[K * 64 + n];
}

// ---------------------------------------------------------------------------
// Encoder stage 1: h1 = relu(x @ W1 + b1), x is [N,2] -> bf16 [N,64]
// ---------------------------------------------------------------------------
__global__ void k_encoder1(const float* __restrict__ x, const float* __restrict__ w1,
                           const float* __restrict__ b1, bf16_t* __restrict__ h1, int n) {
    int idx = blockIdx.x * blockDim.x + threadIdx.x;
    if (idx >= n * 64) return;
    int i = idx >> 6, k = idx & 63;
    float v = x[i * 2 + 0] * w1[k] + x[i * 2 + 1] * w1[64 + k] + b1[k];
    h1[idx] = (bf16_t)(v > 0.0f ? v : 0.0f);
}

// ---------------------------------------------------------------------------
// WMMA GEMM: out[N,64] (f32) = A[N,64] (bf16, row-major) x Wp (packed 64x64 bf16)
// One wave per 16x16 output tile, K=64 -> two chained v_wmma_f32_16x16x32_bf16.
// Block = 256 threads = 8 waves = 2 row tiles x 4 col tiles.
// Optional fused second output: acc2[r] = val * dis[r]^2 (GCN self-loop init).
// ---------------------------------------------------------------------------
__global__ __launch_bounds__(256) void k_gemm64_wmma(
        const bf16_t* __restrict__ A, const bf16_t* __restrict__ Wp,
        const float* __restrict__ bias, int relu,
        float* __restrict__ outf,
        float* __restrict__ acc2, const float* __restrict__ dis,
        int nrows) {
    int wave = threadIdx.x >> 5;
    int lane = threadIdx.x & 31;
    int rowTile = blockIdx.x * 2 + (wave >> 2);
    int nb = wave & 3;
    if (rowTile * 16 >= nrows) return;              // wave-uniform exit

    int m = lane & 15, hi = lane >> 4;
    int arow = rowTile * 16 + m;
    if (arow >= nrows) arow = nrows - 1;            // clamp: EXEC stays all-1 for WMMA

    // A fragments: lane holds K = hi*8..hi*8+7 / 16+hi*8.. (elems 0-7 / 8-15)
    const v8bf* ap = (const v8bf*)(A + (size_t)arow * 64);
    v8bf a00 = ap[0 + hi];      // K  0..15 region
    v8bf a01 = ap[2 + hi];      // K 16..31 region
    v8bf a10 = ap[4 + hi];      // K 32..47 region
    v8bf a11 = ap[6 + hi];      // K 48..63 region
    v16bf a0 = __builtin_shufflevector(a00, a01, 0,1,2,3,4,5,6,7,8,9,10,11,12,13,14,15);
    v16bf a1 = __builtin_shufflevector(a10, a11, 0,1,2,3,4,5,6,7,8,9,10,11,12,13,14,15);

    // B fragments: contiguous 32B per lane from the packed buffer
    const v8bf* bp = (const v8bf*)(Wp + ((size_t)(nb * 2) * 512 + (size_t)lane * 16));
    v8bf b00 = bp[0],  b01 = bp[1];                 // fragment kb=0
    v8bf b10 = bp[64], b11 = bp[65];                // fragment kb=1 (+512 elems)
    v16bf b0 = __builtin_shufflevector(b00, b01, 0,1,2,3,4,5,6,7,8,9,10,11,12,13,14,15);
    v16bf b1 = __builtin_shufflevector(b10, b11, 0,1,2,3,4,5,6,7,8,9,10,11,12,13,14,15);

    v8f c = {};
    c = __builtin_amdgcn_wmma_f32_16x16x32_bf16(false, a0, false, b0, (short)0, c, false, false);
    c = __builtin_amdgcn_wmma_f32_16x16x32_bf16(false, a1, false, b1, (short)0, c, false, false);

    int ncol = nb * 16 + (lane & 15);
    float badd = bias[ncol];
    int r0 = rowTile * 16 + hi * 8;                 // C/D layout: M = hi*8 + v
    float* op = outf + (size_t)r0 * 64 + ncol;

    if (r0 + 8 <= nrows) {                          // full tile: unconditional stores
        float val[8];
#pragma unroll
        for (int v = 0; v < 8; ++v) {
            float t = c[v] + badd;
            if (relu && t < 0.0f) t = 0.0f;
            val[v] = t;
            op[(size_t)v * 64] = t;
        }
        if (acc2) {
            float* ap2 = acc2 + (size_t)r0 * 64 + ncol;
#pragma unroll
            for (int v = 0; v < 8; ++v) {
                float s = dis[r0 + v];
                ap2[(size_t)v * 64] = val[v] * s * s;
            }
        }
    } else {                                        // ragged tail
#pragma unroll
        for (int v = 0; v < 8; ++v) {
            if (r0 + v < nrows) {
                float t = c[v] + badd;
                if (relu && t < 0.0f) t = 0.0f;
                op[(size_t)v * 64] = t;
                if (acc2) {
                    float s = dis[r0 + v];
                    acc2[(size_t)(r0 + v) * 64 + ncol] = t * s * s;
                }
            }
        }
    }
}

// ---------------------------------------------------------------------------
// Edge scatter: one wave per edge, lane = channel; acc lives in 192MB L2
// ---------------------------------------------------------------------------
__global__ __launch_bounds__(256) void k_scatter_edges(
        const int* __restrict__ row, const int* __restrict__ col,
        const float* __restrict__ nrm, const float* __restrict__ hw,
        float* __restrict__ acc, int E) {
    int e = blockIdx.x * 8 + (threadIdx.x >> 5);
    if (e >= E) return;
    int lane = threadIdx.x & 31;
    int r = row[e], c = col[e];
    float nv = nrm[e];
    float v0 = hw[(size_t)r * 64 + lane];
    float v1 = hw[(size_t)r * 64 + 32 + lane];
    atomicAdd(&acc[(size_t)c * 64 + lane],      v0 * nv);
    atomicAdd(&acc[(size_t)c * 64 + 32 + lane], v1 * nv);
}

// ---------------------------------------------------------------------------
// f32 -> (+bias, relu?) -> bf16 activation buffer
// ---------------------------------------------------------------------------
__global__ void k_cvt(const float* __restrict__ src, const float* __restrict__ bias,
                      int relu, bf16_t* __restrict__ dst, int total) {
    int idx = blockIdx.x * blockDim.x + threadIdx.x;
    if (idx >= total) return;
    float v = src[idx] + bias[idx & 63];
    if (relu && v < 0.0f) v = 0.0f;
    dst[idx] = (bf16_t)v;
}

// ---------------------------------------------------------------------------
// Decoder stage 2: [64]->[2] projection + softmax + weighted residual
// ---------------------------------------------------------------------------
__global__ void k_decoder2(const float* __restrict__ t, const float* __restrict__ w2,
                           const float* __restrict__ b2, const float* __restrict__ x,
                           float* __restrict__ out, int n) {
    int i = blockIdx.x * blockDim.x + threadIdx.x;
    if (i >= n) return;
    const float* tp = t + (size_t)i * 64;
    float a0 = b2[0], a1 = b2[1];
#pragma unroll 8
    for (int k = 0; k < 64; ++k) {
        float v = tp[k];
        a0 += v * w2[k * 2 + 0];
        a1 += v * w2[k * 2 + 1];
    }
    float mx = fmaxf(a0, a1);
    float e0 = __expf(a0 - mx), e1 = __expf(a1 - mx);
    float s = e0 + e1;
    out[i * 2 + 0] = e0 / s + 2.0f * x[i * 2 + 0];
    out[i * 2 + 1] = e1 / s;                        // class weight 0 on channel 1
}

// ---------------------------------------------------------------------------
extern "C" void kernel_launch(void* const* d_in, const int* in_sizes, int n_in,
                              void* d_out, int out_size, void* d_ws, size_t ws_size,
                              hipStream_t stream) {
    const float* x      = (const float*)d_in[0];
    const int*   ei     = (const int*)  d_in[1];
    const float* ew     = (const float*)d_in[2];
    const float* enc_w1 = (const float*)d_in[3];
    const float* enc_b1 = (const float*)d_in[4];
    const float* enc_w2 = (const float*)d_in[5];
    const float* enc_b2 = (const float*)d_in[6];
    const float* gcn_w  = (const float*)d_in[7];
    const float* gcn_b  = (const float*)d_in[8];
    const float* dec_w1 = (const float*)d_in[9];
    const float* dec_b1 = (const float*)d_in[10];
    const float* dec_w2 = (const float*)d_in[11];
    const float* dec_b2 = (const float*)d_in[12];

    const int N = in_sizes[0] / 2;
    const int E = in_sizes[2];
    const int L = in_sizes[8] / 64;
    const int* row = ei;
    const int* col = ei + E;
    const int N64 = N * 64;

    // workspace carve-up (256B aligned)
    char* ws = (char*)d_ws;
    size_t off = 0;
    auto carve = [&](size_t bytes) -> char* {
        char* p = ws + off;
        off += (bytes + 255) & ~(size_t)255;
        return p;
    };
    float*  dis   = (float*) carve((size_t)N * 4);        // deg, then rsqrt in place
    float*  nrm   = (float*) carve((size_t)E * 4);
    bf16_t* hbf   = (bf16_t*)carve((size_t)N64 * 2);      // bf16 activations
    float*  hw    = (float*) carve((size_t)N64 * 4);      // GEMM output
    float*  acc   = (float*) carve((size_t)N64 * 4);      // scatter accumulator
    bf16_t* wp    = (bf16_t*)carve((size_t)(L + 2) * 4096 * 2);  // packed weights
    float*  zero64 = (float*)carve(64 * 4);               // zero bias

    const int B = 256;
    int gN   = (N + B - 1) / B;
    int gE   = (E + B - 1) / B;
    int gN64 = (N64 + B - 1) / B;
    int nrt  = (N + 15) / 16;
    int gGemm = (nrt + 1) / 2;
    int gScat = (E + 7) / 8;

    // --- gcn_norm -----------------------------------------------------------
    k_fill_deg  <<<gN, B, 0, stream>>>(dis, N);
    k_acc_deg   <<<gE, B, 0, stream>>>(col, ew, dis, E);
    k_deg_to_dis<<<gN, B, 0, stream>>>(dis, N);
    k_edge_norm <<<gE, B, 0, stream>>>(row, col, ew, dis, nrm, E);
    k_zero64    <<<1, 64, 0, stream>>>(zero64);

    // --- pack weight matrices into WMMA fragment order ----------------------
    k_pack_w<<<16, B, 0, stream>>>(enc_w2, wp);
    for (int l = 0; l < L; ++l)
        k_pack_w<<<16, B, 0, stream>>>(gcn_w + (size_t)l * 4096, wp + (size_t)(1 + l) * 4096);
    k_pack_w<<<16, B, 0, stream>>>(dec_w1, wp + (size_t)(L + 1) * 4096);

    // --- encoder ------------------------------------------------------------
    k_encoder1<<<gN64, B, 0, stream>>>(x, enc_w1, enc_b1, hbf, N);
    k_gemm64_wmma<<<gGemm, B, 0, stream>>>(hbf, wp, enc_b2, 0, hw, nullptr, nullptr, N);
    k_cvt<<<gN64, B, 0, stream>>>(hw, zero64, 0, hbf, N64);

    // --- GCN layers (GEMM fuses self-loop init of acc) ----------------------
    for (int l = 0; l < L; ++l) {
        k_gemm64_wmma <<<gGemm, B, 0, stream>>>(hbf, wp + (size_t)(1 + l) * 4096,
                                                zero64, 0, hw, acc, dis, N);
        k_scatter_edges<<<gScat, B, 0, stream>>>(row, col, nrm, hw, acc, E);
        k_cvt<<<gN64, B, 0, stream>>>(acc, gcn_b + (size_t)l * 64, 1, hbf, N64);
    }

    // --- decoder ------------------------------------------------------------
    k_gemm64_wmma<<<gGemm, B, 0, stream>>>(hbf, wp + (size_t)(L + 1) * 4096,
                                           dec_b1, 1, hw, nullptr, nullptr, N);
    k_decoder2<<<gN, B, 0, stream>>>(hw, dec_w2, dec_b2, x, (float*)d_out, N);
}
